// MoEConnectionProcessor_78305843741404
// MI455X (gfx1250) — matrix operand order repaired and checked by reading, not compiled, not run on hardware
//
#include <hip/hip_runtime.h>

// ---------------------------------------------------------------------------
// MoE connection processor for MI455X (gfx1250, wave32, WMMA 16x16x32 f16).
// 1 wave = 16 cells. msgs GEMM + all concat layers run on v_wmma_f32_16x16x32_f16
// with f32 accumulation; tier masks via exact integer distance^2 thresholds.
// v2: all divisions -> v_rcp_f32; tanh -> v_tanh_f32 if declared, else
//     v_exp_f32 (exp2) + v_rcp_f32.
// ---------------------------------------------------------------------------

typedef __attribute__((ext_vector_type(16))) _Float16 v16h;
typedef __attribute__((ext_vector_type(8)))  float    v8f;

#define LATT_N 262144
#define LATT_K 26
#define LATT_S 32
#define CNF_DT 0.3f

// per-wave LDS region layout (bytes)
#define W_STAGE   0      // 32x32 f16 neighbor staging   (2048 B)
#define W_XS      2048   // 16x32 f16 x                  (1024 B)
#define W_MLOC    3072   // 16x32 f16 mean_local         (1024 B)
#define W_MDIST   4096   // 16x32 f16 mean_dist          (1024 B)
#define W_MALL    5120   // 16x32 f16 mean_all           (1024 B)
#define W_MFUNC   6144   // 16x32 f16 msg_func           (1024 B)
#define W_LOCALO  7168   // 16x32 f32 local_out          (2048 B)
#define W_HF      9216   // 16x32 f32 h (functional CNF) (2048 B)
#define W_HD      11264  // 16x32 f32 h (distant CNF)    (2048 B)
#define W_G1O     13312  // 16x16 f32 gating hidden      (1024 B)
#define W_GATES   14336  // 16x4  f32 gates              ( 256 B)
#define WAVE_LDS  14592

// shared weight region (f16), sizes in halves
#define WB_MSG 0       // 32*32
#define WB_LOC 1024    // 64*32
#define WB_CF  3072    // 64*32
#define WB_CD  5120    // 64*32
#define WB_G1  7168    // 64*16
#define SMEM_BYTES (16384 + 8 * WAVE_LDS)

__device__ __forceinline__ float fast_rcp(float x) {
  return __builtin_amdgcn_rcpf(x);       // v_rcp_f32, ~1ulp
}

__device__ __forceinline__ float fast_tanh(float x) {
#if __has_builtin(__builtin_amdgcn_tanhf)
  return __builtin_amdgcn_tanhf(x);      // v_tanh_f32 (CDNA5 TRANS op)
#elif __has_builtin(__builtin_amdgcn_tanh_f32)
  return __builtin_amdgcn_tanh_f32(x);
#else
  // tanh(x) = 1 - 2/(exp2(2*log2e*x)+1); v_exp_f32 + v_rcp_f32.
  // Saturates correctly: exp2->inf => 1, exp2->0 => -1.
  float e = __builtin_amdgcn_exp2f(x * (2.0f * 1.4426950408889634f));
  return 1.0f - 2.0f * fast_rcp(e + 1.0f);
#endif
}

__device__ __forceinline__ v8f wmma32(v16h a, v16h b, v8f c) {
  return __builtin_amdgcn_wmma_f32_16x16x32_f16(false, a, false, b,
                                                (short)0, c, false, false);
}

// A-tile (16x32, f16) from an LDS f16 array with row stride 32.
// Lane L holds row M = L%16; K packing per CDNA5 ISA 16-bit A layout.
__device__ __forceinline__ v16h load_a_h(const _Float16* src, int rbase, int lane) {
  const int row  = rbase + (lane & 15);
  const int half = lane >> 4;
  v16h a;
#pragma unroll
  for (int j = 0; j < 8; ++j) {
    const int k0 = 2 * (j & 3) + ((j >= 4) ? 16 : 0) + (half ? 8 : 0);
    a[2 * j]     = src[row * 32 + k0];
    a[2 * j + 1] = src[row * 32 + k0 + 1];
  }
  return a;
}

// A-tile from an LDS f32 array (converted to f16 on load).
__device__ __forceinline__ v16h load_a_f32(const float* src, int lane) {
  const int row  = lane & 15;
  const int half = lane >> 4;
  v16h a;
#pragma unroll
  for (int j = 0; j < 8; ++j) {
    const int k0 = 2 * (j & 3) + ((j >= 4) ? 16 : 0) + (half ? 8 : 0);
    a[2 * j]     = (_Float16)src[row * 32 + k0];
    a[2 * j + 1] = (_Float16)src[row * 32 + k0 + 1];
  }
  return a;
}

// B-tile (32x16, f16) from row-major f16 weights [Krows][ldw].
// Lane L holds column N = L%16 + 16*coltile; lanes 0-15 cover K rows
// +0..7,+16..23, lanes 16-31 cover +8..15,+24..31 (2 consecutive K per VGPR).
__device__ __forceinline__ v16h load_b(const _Float16* w, int ldw,
                                       int krow_base, int coltile, int lane) {
  const int n    = (lane & 15) + 16 * coltile;
  const int half = lane >> 4;
  v16h b;
#pragma unroll
  for (int j = 0; j < 8; ++j) {
    const int k = krow_base + 2 * (j & 3) + ((j >= 4) ? 16 : 0) + (half ? 8 : 0);
    b[2 * j]     = w[k * ldw + n];
    b[2 * j + 1] = w[(k + 1) * ldw + n];
  }
  return b;
}

__global__ __launch_bounds__(256)
void moe_conn_kernel(const float* __restrict__ states,
                     const float* __restrict__ Wloc_g,  const float* __restrict__ bloc,
                     const float* __restrict__ Wmsg_g,  const float* __restrict__ bmsg,
                     const float* __restrict__ Wcf_g,   const float* __restrict__ bcf,
                     const float* __restrict__ Wcd_g,   const float* __restrict__ bcd,
                     const float* __restrict__ Wg1_g,   const float* __restrict__ bg1,
                     const float* __restrict__ Wg2_g,   const float* __restrict__ bg2,
                     const int*   __restrict__ nidx,
                     float* __restrict__ out) {
  extern __shared__ char smem[];
  _Float16* wh = (_Float16*)smem;  // shared f16 weight copies

  const int tid  = threadIdx.x;
  const int lane = tid & 31;
  const int wv   = tid >> 5;

  // ---- block-cooperative weight conversion to f16 in LDS ----
  for (int i = tid; i < 1024; i += 256) wh[WB_MSG + i] = (_Float16)Wmsg_g[i];
  for (int i = tid; i < 2048; i += 256) wh[WB_LOC + i] = (_Float16)Wloc_g[i];
  for (int i = tid; i < 2048; i += 256) wh[WB_CF  + i] = (_Float16)Wcf_g[i];
  for (int i = tid; i < 2048; i += 256) wh[WB_CD  + i] = (_Float16)Wcd_g[i];
  for (int i = tid; i < 1024; i += 256) wh[WB_G1  + i] = (_Float16)Wg1_g[i];
  __syncthreads();

  // ---- per-wave LDS region ----
  char* wb = smem + 16384 + wv * WAVE_LDS;
  _Float16* stage = (_Float16*)(wb + W_STAGE);
  _Float16* xs    = (_Float16*)(wb + W_XS);
  _Float16* mloc  = (_Float16*)(wb + W_MLOC);
  _Float16* mdist = (_Float16*)(wb + W_MDIST);
  _Float16* mall  = (_Float16*)(wb + W_MALL);
  _Float16* mfunc = (_Float16*)(wb + W_MFUNC);
  float* localo = (float*)(wb + W_LOCALO);
  float* hf     = (float*)(wb + W_HF);
  float* hd     = (float*)(wb + W_HD);
  float* g1o    = (float*)(wb + W_G1O);
  float* gates  = (float*)(wb + W_GATES);

  const int cellbase = (blockIdx.x * 8 + wv) * 16;
  const int half = lane >> 4;
  const int nlo  = lane & 15;

  // zero pad staging rows 26..31 once (never written by gather loop)
#pragma unroll
  for (int r = LATT_K; r < 32; ++r) stage[r * 32 + lane] = (_Float16)0.0f;

  // W_msg B-tiles: invariant across cells
  const v16h Bm0 = load_b(wh + WB_MSG, 32, 0, 0, lane);
  const v16h Bm1 = load_b(wh + WB_MSG, 32, 0, 1, lane);
  const float bm0 = bmsg[nlo];
  const float bm1 = bmsg[16 + nlo];

  // =================== Phase A+B: gather, tier stats, msgs GEMM ============
  for (int c = 0; c < 16; ++c) {
    const int cell = cellbase + c;
    const int cx = cell >> 12, cy = (cell >> 6) & 63, cz = cell & 63;

    const float vx = states[cell * LATT_S + lane];  // lane = channel
    xs[c * 32 + lane] = (_Float16)vx;
    hf[c * 32 + lane] = vx;
    hd[c * 32 + lane] = vx;

    float sa = 0.f, sl = 0.f, sd = 0.f;
    int cl = 0, cf = 0, cdd = 0;
    unsigned fbits = 0u;

#pragma unroll 2
    for (int k = 0; k < LATT_K; ++k) {
      const int idx = nidx[cell * LATT_K + k];
      const float nv = states[idx * LATT_S + lane];     // coalesced 128B row
      const int nx = idx >> 12, ny = (idx >> 6) & 63, nz = idx & 63;
      const int dx = nx - cx, dy = ny - cy, dz = nz - cz;
      const int d2 = dx * dx + dy * dy + dz * dz;       // exact integer dist^2
      sa += nv;
      if (d2 <= 3)               { sl += nv; ++cl; }                 // r<=1.8
      else if (d2 <= 12)         { fbits |= (1u << k); ++cf; }       // r<=3.5
      else                       { sd += nv; ++cdd; }
      stage[k * 32 + lane] = (_Float16)nv;
    }

    mall [c * 32 + lane] = (_Float16)(sa * (1.0f / 26.0f));
    mloc [c * 32 + lane] = (_Float16)(sl * fast_rcp(fmaxf((float)cl, 1.0f)));
    mdist[c * 32 + lane] = (_Float16)(sd * fast_rcp(fmaxf((float)cdd, 1.0f)));

    // msgs = tanh(nbr @ W_msg + b_msg), masked-mean over functional tier
    float acc0 = 0.f, acc1 = 0.f;
#pragma unroll
    for (int t = 0; t < 2; ++t) {
      const v16h A = load_a_h(stage, 16 * t, lane);
      v8f z = {};
      v8f d0 = wmma32(A, Bm0, z);
      v8f d1 = wmma32(A, Bm1, z);
#pragma unroll
      for (int v = 0; v < 8; ++v) {
        const int nb = 16 * t + v + 8 * half;            // neighbor (row) id
        const float mk = (fbits >> nb) & 1u ? 1.0f : 0.0f;
        acc0 += mk * fast_tanh(d0[v] + bm0);
        acc1 += mk * fast_tanh(d1[v] + bm1);
      }
    }
    acc0 += __shfl_xor(acc0, 16, 32);
    acc1 += __shfl_xor(acc1, 16, 32);
    const float inv_cf = fast_rcp(fmaxf((float)cf, 1.0f));
    // lanes 0-15 hold cols 0-15 (tile0); lanes 16-31 hold cols 16-31 (tile1)
    mfunc[c * 32 + lane] = (_Float16)(((lane < 16) ? acc0 : acc1) * inv_cf);
  }

  // =================== Phase C: batched 16-cell concat layers ==============
  // --- local expert: tanh([x, mean_local] @ W_local + b_local) ---
  {
    const v16h A0 = load_a_h(xs, 0, lane);
    const v16h A1 = load_a_h(mloc, 0, lane);
    const v16h B00 = load_b(wh + WB_LOC, 32, 0,  0, lane);
    const v16h B01 = load_b(wh + WB_LOC, 32, 0,  1, lane);
    const v16h B10 = load_b(wh + WB_LOC, 32, 32, 0, lane);
    const v16h B11 = load_b(wh + WB_LOC, 32, 32, 1, lane);
    v8f z = {};
    v8f d0 = wmma32(A0, B00, z); d0 = wmma32(A1, B10, d0);
    v8f d1 = wmma32(A0, B01, z); d1 = wmma32(A1, B11, d1);
    const float bb0 = bloc[nlo], bb1 = bloc[16 + nlo];
#pragma unroll
    for (int v = 0; v < 8; ++v) {
      const int m = v + 8 * half;
      localo[m * 32 + nlo]      = fast_tanh(d0[v] + bb0);
      localo[m * 32 + nlo + 16] = fast_tanh(d1[v] + bb1);
    }
  }

  // --- functional expert: 3-step Euler CNF on [h, msg_func] ---
  {
    const v16h Am  = load_a_h(mfunc, 0, lane);
    const v16h B00 = load_b(wh + WB_CF, 32, 0,  0, lane);
    const v16h B01 = load_b(wh + WB_CF, 32, 0,  1, lane);
    const v16h B10 = load_b(wh + WB_CF, 32, 32, 0, lane);
    const v16h B11 = load_b(wh + WB_CF, 32, 32, 1, lane);
    const float bb0 = bcf[nlo], bb1 = bcf[16 + nlo];
    for (int step = 0; step < 3; ++step) {
      const v16h Ah = load_a_f32(hf, lane);
      v8f z = {};
      v8f d0 = wmma32(Ah, B00, z); d0 = wmma32(Am, B10, d0);
      v8f d1 = wmma32(Ah, B01, z); d1 = wmma32(Am, B11, d1);
#pragma unroll
      for (int v = 0; v < 8; ++v) {
        const int m = v + 8 * half;
        hf[m * 32 + nlo]      += CNF_DT * fast_tanh(d0[v] + bb0);
        hf[m * 32 + nlo + 16] += CNF_DT * fast_tanh(d1[v] + bb1);
      }
    }
  }

  // --- distant expert: 3-step Euler CNF on [h, mean_dist] ---
  {
    const v16h Am  = load_a_h(mdist, 0, lane);
    const v16h B00 = load_b(wh + WB_CD, 32, 0,  0, lane);
    const v16h B01 = load_b(wh + WB_CD, 32, 0,  1, lane);
    const v16h B10 = load_b(wh + WB_CD, 32, 32, 0, lane);
    const v16h B11 = load_b(wh + WB_CD, 32, 32, 1, lane);
    const float bb0 = bcd[nlo], bb1 = bcd[16 + nlo];
    for (int step = 0; step < 3; ++step) {
      const v16h Ah = load_a_f32(hd, lane);
      v8f z = {};
      v8f d0 = wmma32(Ah, B00, z); d0 = wmma32(Am, B10, d0);
      v8f d1 = wmma32(Ah, B01, z); d1 = wmma32(Am, B11, d1);
#pragma unroll
      for (int v = 0; v < 8; ++v) {
        const int m = v + 8 * half;
        hd[m * 32 + nlo]      += CNF_DT * fast_tanh(d0[v] + bb0);
        hd[m * 32 + nlo + 16] += CNF_DT * fast_tanh(d1[v] + bb1);
      }
    }
  }

  // --- gating hidden: tanh([x, mean_all] @ W_g1 + b_g1), 16 outputs ---
  {
    const v16h A0 = load_a_h(xs, 0, lane);
    const v16h A1 = load_a_h(mall, 0, lane);
    const v16h B0 = load_b(wh + WB_G1, 16, 0,  0, lane);  // ldw = 16
    const v16h B1 = load_b(wh + WB_G1, 16, 32, 0, lane);
    v8f z = {};
    v8f d0 = wmma32(A0, B0, z); d0 = wmma32(A1, B1, d0);
    const float bb = bg1[nlo];
#pragma unroll
    for (int v = 0; v < 8; ++v) {
      const int m = v + 8 * half;
      g1o[m * 16 + nlo] = fast_tanh(d0[v] + bb);
    }
  }

  // --- gating output + softmax: lane = cell (lanes 0-15 commit) ---
  {
    const int cr = nlo;
    float l0 = bg2[0], l1 = bg2[1], l2 = bg2[2];
#pragma unroll
    for (int j = 0; j < 16; ++j) {
      const float gv = g1o[cr * 16 + j];
      l0 += gv * Wg2_g[j * 3 + 0];
      l1 += gv * Wg2_g[j * 3 + 1];
      l2 += gv * Wg2_g[j * 3 + 2];
    }
    const float mx = fmaxf(l0, fmaxf(l1, l2));
    const float k2 = 1.4426950408889634f;
    const float e0 = __builtin_amdgcn_exp2f((l0 - mx) * k2);
    const float e1 = __builtin_amdgcn_exp2f((l1 - mx) * k2);
    const float e2 = __builtin_amdgcn_exp2f((l2 - mx) * k2);
    const float inv = fast_rcp(e0 + e1 + e2);
    if (lane < 16) {
      gates[cr * 4 + 0] = e0 * inv;
      gates[cr * 4 + 1] = e1 * inv;
      gates[cr * 4 + 2] = e2 * inv;
    }
  }

  // --- weighted combine & coalesced store (lane = channel) ---
#pragma unroll 4
  for (int r = 0; r < 16; ++r) {
    const float g0 = gates[r * 4 + 0];
    const float g1 = gates[r * 4 + 1];
    const float g2 = gates[r * 4 + 2];
    const float o = g0 * localo[r * 32 + lane]
                  + g1 * hf[r * 32 + lane]
                  + g2 * hd[r * 32 + lane];
    out[(cellbase + r) * LATT_S + lane] = o;
  }
}

extern "C" void kernel_launch(void* const* d_in, const int* in_sizes, int n_in,
                              void* d_out, int out_size, void* d_ws, size_t ws_size,
                              hipStream_t stream) {
  (void)n_in; (void)out_size; (void)d_ws; (void)ws_size;
  const float* states = (const float*)d_in[0];
  const float* Wloc   = (const float*)d_in[1];
  const float* bloc   = (const float*)d_in[2];
  const float* Wmsg   = (const float*)d_in[3];
  const float* bmsg   = (const float*)d_in[4];
  const float* Wcf    = (const float*)d_in[5];
  const float* bcf    = (const float*)d_in[6];
  const float* Wcd    = (const float*)d_in[7];
  const float* bcd    = (const float*)d_in[8];
  const float* Wg1    = (const float*)d_in[9];
  const float* bg1    = (const float*)d_in[10];
  const float* Wg2    = (const float*)d_in[11];
  const float* bg2    = (const float*)d_in[12];
  const int*   nidx   = (const int*)d_in[13];
  float* out = (float*)d_out;

  const int n_cells = in_sizes[0] / LATT_S;       // 262144
  const int blocks  = n_cells / (8 * 16);         // 8 waves * 16 cells / block

  moe_conn_kernel<<<blocks, 256, SMEM_BYTES, stream>>>(
      states, Wloc, bloc, Wmsg, bmsg, Wcf, bcf, Wcd, bcd,
      Wg1, bg1, Wg2, bg2, nidx, out);
}
